// MEGABlock_88716844466541
// MI455X (gfx1250) — matrix operand order repaired and checked by reading, not compiled, not run on hardware
//
#include <hip/hip_runtime.h>
#include <hip/hip_bf16.h>

typedef __attribute__((ext_vector_type(16))) __bf16         v16bf;
typedef __attribute__((ext_vector_type(8)))  float          v8f;
typedef __attribute__((ext_vector_type(8)))  unsigned short ush8;
typedef __attribute__((ext_vector_type(16))) unsigned short ush16;

#define DM 512          // d_model
#define DA 256          // d_attn
#define DF 1024         // d_ffn
#define DST 16          // d_state
#define BB 8
#define LL 2048
#define NTOK (BB*LL)    // 16384 tokens

// ---------- helpers ----------
static __device__ __forceinline__ unsigned short bfbits(float f) {
  unsigned u = __builtin_bit_cast(unsigned, f);
  u += 0x7FFFu + ((u >> 16) & 1u);          // round-to-nearest-even
  return (unsigned short)(u >> 16);
}
static __device__ __forceinline__ v16bf pack16(ush8 lo, ush8 hi) {
  ush16 u;
#pragma unroll
  for (int i = 0; i < 8; ++i) { u[i] = lo[i]; u[i + 8] = hi[i]; }
  return __builtin_bit_cast(v16bf, u);
}
static __device__ __forceinline__ v8f wmma_bf16(v16bf a, v16bf b, v8f c) {
  return __builtin_amdgcn_wmma_f32_16x16x32_bf16(false, a, false, b, (short)0, c,
                                                 false, false);
}
static __device__ __forceinline__ void wait_ds0() {
#if __has_builtin(__builtin_amdgcn_s_wait_dscnt)
  __builtin_amdgcn_s_wait_dscnt(0);
#else
  asm volatile("s_wait_dscnt 0" ::: "memory");
#endif
}

// ---------- fp32 -> bf16 weight conversion ----------
__global__ void cvt_kernel(const float* __restrict__ src,
                           unsigned short* __restrict__ dst, int n) {
  int i = blockIdx.x * blockDim.x + threadIdx.x;
  if (i < n) dst[i] = bfbits(src[i]);
}

// ---------- LayerNorm over D=512, one block per token ----------
__global__ __launch_bounds__(128) void ln_kernel(
    const float* __restrict__ x, const float* __restrict__ w,
    const float* __restrict__ bsh, float* __restrict__ outf,
    unsigned short* __restrict__ outb) {
  __shared__ float red[128];
  const int t = threadIdx.x;
  const size_t base = (size_t)blockIdx.x * DM;
  float v[4];
  float s = 0.f;
#pragma unroll
  for (int i = 0; i < 4; ++i) { v[i] = x[base + t + i * 128]; s += v[i]; }
  red[t] = s; __syncthreads();
  for (int st = 64; st > 0; st >>= 1) { if (t < st) red[t] += red[t + st]; __syncthreads(); }
  const float mean = red[0] * (1.f / DM);
  __syncthreads();
  float s2 = 0.f;
#pragma unroll
  for (int i = 0; i < 4; ++i) { float d = v[i] - mean; s2 += d * d; }
  red[t] = s2; __syncthreads();
  for (int st = 64; st > 0; st >>= 1) { if (t < st) red[t] += red[t + st]; __syncthreads(); }
  const float inv = rsqrtf(red[0] * (1.f / DM) + 1e-5f);
#pragma unroll
  for (int i = 0; i < 4; ++i) {
    const int c = t + i * 128;
    const float o = (v[i] - mean) * inv * w[c] + bsh[c];
    if (outf) outf[base + c] = o;
    if (outb) outb[base + c] = bfbits(o);
  }
}

// ---------- EMA scan: one lane per (b,d) chain, 16-state in registers ----------
__global__ void ema_kernel(const float* __restrict__ xn,
                           const float* __restrict__ ap,
                           const float* __restrict__ dp,
                           const float* __restrict__ gm,
                           const float* __restrict__ bt,
                           unsigned short* __restrict__ ybf) {
  const int tid = blockIdx.x * blockDim.x + threadIdx.x;
  if (tid >= BB * DM) return;
  const int b = tid / DM, d = tid % DM;
  float al[DST], om[DST], de[DST], h[DST];
#pragma unroll
  for (int s = 0; s < DST; ++s) {
    al[s] = 1.f / (1.f + __expf(-ap[d * DST + s]));
    om[s] = 1.f - al[s];
    de[s] = 1.f / (1.f + __expf(-dp[d * DST + s]));
    h[s] = 0.f;
  }
  const float g = gm[d], be = bt[d];
  const float* xp = xn + (size_t)b * LL * DM + d;
  unsigned short* yp = ybf + (size_t)b * LL * DM + d;
  for (int l = 0; l < LL; ++l) {
    const float xt = xp[(size_t)l * DM];
    float y = 0.f;
#pragma unroll
    for (int s = 0; s < DST; ++s) { h[s] = al[s] * h[s] + om[s] * xt; y += h[s] * de[s]; }
    yp[(size_t)l * DM] = bfbits(y * g + be);
  }
}

// ---------- bf16 WMMA GEMM: C[M,N] = A[M,K]@W[K,N] (+bias, act, resid) ----------
// block = 4 waves; wave -> 16x64 tile; W chunk staged transposed in LDS.
__global__ __launch_bounds__(128) void gemm_bf16_kernel(
    const unsigned short* __restrict__ A, const unsigned short* __restrict__ W,
    const float* __restrict__ bias, const float* __restrict__ resid,
    void* __restrict__ out, int M, int K, int N, int outBf16, int act) {
  __shared__ __align__(16) unsigned short Bt[64 * 32];  // [col][k]
  const int lane = threadIdx.x & 31;
  const int wave = threadIdx.x >> 5;
  const int rt = (blockIdx.x * 4 + wave) * 16;
  const int cb = blockIdx.y * 64;
  v8f acc[4];
#pragma unroll
  for (int nt = 0; nt < 4; ++nt)
#pragma unroll
    for (int r = 0; r < 8; ++r) acc[nt][r] = 0.f;

  const int wr = threadIdx.x >> 2;  // 0..31 : k row within chunk
  const int wq = threadIdx.x & 3;   // 0..3  : 16-col group
  const int arow = rt + (lane & 15);
  const int koffA = (lane >> 4) * 8;
  const int koffB = (lane >> 4) * 8;

  for (int kc = 0; kc < K; kc += 32) {
    const unsigned short* wp = W + (size_t)(kc + wr) * N + cb + wq * 16;
    ush8 w0 = *(const ush8*)(wp);
    ush8 w1 = *(const ush8*)(wp + 8);
    __syncthreads();  // previous-iteration readers done
#pragma unroll
    for (int j = 0; j < 8; ++j) {
      Bt[(wq * 16 + j) * 32 + wr]     = w0[j];
      Bt[(wq * 16 + 8 + j) * 32 + wr] = w1[j];
    }
    __syncthreads();
    const unsigned short* ap = A + (size_t)arow * K + kc + koffA;
    v16bf av = pack16(*(const ush8*)ap, *(const ush8*)(ap + 16));
    __builtin_prefetch(ap + 32, 0, 0);
#pragma unroll
    for (int nt = 0; nt < 4; ++nt) {
      const unsigned short* bp = &Bt[(nt * 16 + (lane & 15)) * 32 + koffB];
      v16bf bv = pack16(*(const ush8*)bp, *(const ush8*)(bp + 16));
      acc[nt] = wmma_bf16(av, bv, acc[nt]);
    }
  }
#pragma unroll
  for (int nt = 0; nt < 4; ++nt) {
    const int col = cb + nt * 16 + (lane & 15);
    const float bv = bias ? bias[col] : 0.f;
#pragma unroll
    for (int r = 0; r < 8; ++r) {
      const int row = rt + r + ((lane >> 4) * 8);
      float v = acc[nt][r] + bv;
      if (act == 1)      v = 1.f / (1.f + __expf(-v));
      else if (act == 2) v = 0.5f * v * (1.f + erff(v * 0.70710678118f));
      if (resid) v += resid[(size_t)row * N + col];
      if (outBf16) ((unsigned short*)out)[(size_t)row * N + col] = bfbits(v);
      else         ((float*)out)[(size_t)row * N + col] = v;
    }
  }
}

// ---------- flash attention + gate: block = (b, 16-query tile), 4 waves x 128 cols ----------
__global__ __launch_bounds__(128) void attn_kernel(
    const unsigned short* __restrict__ Qb, const unsigned short* __restrict__ Kb,
    const unsigned short* __restrict__ Vb, const int* __restrict__ amask,
    const float* __restrict__ G, unsigned short* __restrict__ ctx) {
  __shared__ __align__(16) unsigned short Vt[DM * 32];      // V chunk transposed, 32KB
  __shared__ __align__(16) unsigned short Pt[4 * 16 * 32];  // per-wave P tile, 4KB
  const int lane = threadIdx.x & 31;
  const int wave = threadIdx.x >> 5;
  const int b = blockIdx.y;
  const int qt = blockIdx.x * 16;
  const int colbase = wave * 128;
  const int ln15 = lane & 15;
  const int half = lane >> 4;

  v8f O[8];
  float mrow[8], lrow[8];
#pragma unroll
  for (int nt = 0; nt < 8; ++nt)
#pragma unroll
    for (int r = 0; r < 8; ++r) O[nt][r] = 0.f;
#pragma unroll
  for (int r = 0; r < 8; ++r) { mrow[r] = -1e30f; lrow[r] = 0.f; }

  const unsigned short* qrow = Qb + ((size_t)(b * LL) + qt + ln15) * DA;
  unsigned short* pmine = &Pt[wave * 16 * 32];
  const int vr = threadIdx.x & 31;
  const int vg = threadIdx.x >> 5;

  for (int kt = 0; kt < LL; kt += 32) {
    {  // cooperative stage: V[kt..kt+31][:] -> Vt[col][k]
      const unsigned short* vp = Vb + ((size_t)(b * LL) + kt + vr) * DM + vg * 128;
      ush8 tmp[16];
#pragma unroll
      for (int i = 0; i < 16; ++i) tmp[i] = *(const ush8*)(vp + i * 8);
      __syncthreads();
#pragma unroll
      for (int i = 0; i < 16; ++i)
#pragma unroll
        for (int j = 0; j < 8; ++j)
          Vt[(vg * 128 + i * 8 + j) * 32 + vr] = tmp[i][j];
      __syncthreads();
    }
    float sA[8], sB[8];
    for (int hk = 0; hk < 2; ++hk) {  // two 16-key S tiles
      v8f s;
#pragma unroll
      for (int r = 0; r < 8; ++r) s[r] = 0.f;
      const int key = kt + hk * 16 + ln15;
      const unsigned short* krow = Kb + ((size_t)(b * LL) + key) * DA;
#pragma unroll
      for (int kc = 0; kc < DA; kc += 32) {
        v16bf aq = pack16(*(const ush8*)(qrow + kc + half * 8),
                          *(const ush8*)(qrow + kc + half * 8 + 16));
        v16bf bk = pack16(*(const ush8*)(krow + kc + half * 8),
                          *(const ush8*)(krow + kc + half * 8 + 16));
        s = wmma_bf16(aq, bk, s);
      }
      const int mk = amask[b * LL + key];
      float* dst = hk ? sB : sA;
#pragma unroll
      for (int r = 0; r < 8; ++r) dst[r] = mk ? (s[r] * 0.0625f) : -1e30f;
    }
    float corr[8];
#pragma unroll
    for (int r = 0; r < 8; ++r) {  // online-softmax stats (rows split by lane half)
      float t = fmaxf(sA[r], sB[r]);
#pragma unroll
      for (int m = 1; m < 16; m <<= 1) t = fmaxf(t, __shfl_xor(t, m, 32));
      float mn = fmaxf(mrow[r], t);
      corr[r] = __expf(mrow[r] - mn);
      mrow[r] = mn;
      lrow[r] *= corr[r];
    }
#pragma unroll
    for (int nt = 0; nt < 8; ++nt)
#pragma unroll
      for (int r = 0; r < 8; ++r) O[nt][r] *= corr[r];
#pragma unroll
    for (int r = 0; r < 8; ++r) {
      float pa = __expf(sA[r] - mrow[r]);
      float pb = __expf(sB[r] - mrow[r]);
      const int row = r + half * 8;
      pmine[row * 32 + ln15]      = bfbits(pa);
      pmine[row * 32 + 16 + ln15] = bfbits(pb);
      float rs = pa + pb;
#pragma unroll
      for (int m = 1; m < 16; m <<= 1) rs += __shfl_xor(rs, m, 32);
      lrow[r] += rs;
    }
    wait_ds0();  // in-wave LDS RAW on Pt
    const unsigned short* pr = &pmine[ln15 * 32 + half * 8];
    v16bf pfrag = pack16(*(const ush8*)(pr), *(const ush8*)(pr + 16));
#pragma unroll
    for (int nt = 0; nt < 8; ++nt) {  // P(16x32) x V(32x128 slice)
      const unsigned short* vb = &Vt[(colbase + nt * 16 + ln15) * 32 + half * 8];
      v16bf vfrag = pack16(*(const ush8*)(vb), *(const ush8*)(vb + 16));
      O[nt] = wmma_bf16(pfrag, vfrag, O[nt]);
    }
  }
#pragma unroll
  for (int nt = 0; nt < 8; ++nt) {  // normalize, gate, store bf16 ctx
    const int col = colbase + nt * 16 + ln15;
#pragma unroll
    for (int r = 0; r < 8; ++r) {
      const int q = qt + r + half * 8;
      float v = O[nt][r] / lrow[r];
      float g = G[((size_t)(b * LL) + q) * DM + col];
      ctx[((size_t)(b * LL) + q) * DM + col] = bfbits(v * g);
    }
  }
}

// ---------- host orchestration ----------
extern "C" void kernel_launch(void* const* d_in, const int* in_sizes, int n_in,
                              void* d_out, int out_size, void* d_ws, size_t ws_size,
                              hipStream_t stream) {
  (void)in_sizes; (void)n_in; (void)out_size; (void)ws_size;
  const float* x   = (const float*)d_in[0];
  const int*   am  = (const int*)d_in[1];
  const float* ap  = (const float*)d_in[2];
  const float* dp  = (const float*)d_in[3];
  const float* eg  = (const float*)d_in[4];
  const float* eb  = (const float*)d_in[5];
  const float* l1w = (const float*)d_in[6];
  const float* l1b = (const float*)d_in[7];
  const float* l2w = (const float*)d_in[8];
  const float* l2b = (const float*)d_in[9];
  const float* Wq  = (const float*)d_in[10];
  const float* Wk  = (const float*)d_in[11];
  const float* Wv  = (const float*)d_in[12];
  const float* Wg  = (const float*)d_in[13];
  const float* bg  = (const float*)d_in[14];
  const float* Wo  = (const float*)d_in[15];
  const float* W1  = (const float*)d_in[16];
  const float* b1  = (const float*)d_in[17];
  const float* W2  = (const float*)d_in[18];
  const float* b2  = (const float*)d_in[19];

  char* w = (char*)d_ws;
  size_t off = 0;
  auto alloc = [&](size_t bytes) -> void* {
    void* p = w + off;
    off = (off + bytes + 255) & ~(size_t)255;
    return p;
  };
  float*          xn  = (float*)alloc((size_t)NTOK * DM * 4);
  unsigned short* ybf = (unsigned short*)alloc((size_t)NTOK * DM * 2);
  unsigned short* Qb  = (unsigned short*)alloc((size_t)NTOK * DA * 2);
  unsigned short* Kbf = (unsigned short*)alloc((size_t)NTOK * DA * 2);
  unsigned short* Vbf = (unsigned short*)alloc((size_t)NTOK * DM * 2);
  float*          Gf  = (float*)alloc((size_t)NTOK * DM * 4);
  unsigned short* ctx = (unsigned short*)alloc((size_t)NTOK * DM * 2);
  float*          hf  = (float*)alloc((size_t)NTOK * DM * 4);
  unsigned short* hnb = (unsigned short*)alloc((size_t)NTOK * DM * 2);
  unsigned short* t1b = (unsigned short*)alloc((size_t)NTOK * DF * 2);
  unsigned short* Wqb = (unsigned short*)alloc((size_t)DM * DA * 2);
  unsigned short* Wkb = (unsigned short*)alloc((size_t)DM * DA * 2);
  unsigned short* Wvb = (unsigned short*)alloc((size_t)DM * DM * 2);
  unsigned short* Wgb = (unsigned short*)alloc((size_t)DM * DM * 2);
  unsigned short* Wob = (unsigned short*)alloc((size_t)DM * DM * 2);
  unsigned short* W1b = (unsigned short*)alloc((size_t)DM * DF * 2);
  unsigned short* W2b = (unsigned short*)alloc((size_t)DF * DM * 2);

  auto cv = [&](const float* s, unsigned short* d, int n) {
    cvt_kernel<<<(n + 255) / 256, 256, 0, stream>>>(s, d, n);
  };
  cv(Wq, Wqb, DM * DA);
  cv(Wk, Wkb, DM * DA);
  cv(Wv, Wvb, DM * DM);
  cv(Wg, Wgb, DM * DM);
  cv(Wo, Wob, DM * DM);
  cv(W1, W1b, DM * DF);
  cv(W2, W2b, DF * DM);

  // LN1 -> EMA
  ln_kernel<<<NTOK, 128, 0, stream>>>(x, l1w, l1b, xn, nullptr);
  ema_kernel<<<(BB * DM + 255) / 256, 256, 0, stream>>>(xn, ap, dp, eg, eb, ybf);

  // projections
  gemm_bf16_kernel<<<dim3(NTOK / 64, DA / 64), 128, 0, stream>>>(
      ybf, Wqb, nullptr, nullptr, Qb, NTOK, DM, DA, 1, 0);
  gemm_bf16_kernel<<<dim3(NTOK / 64, DA / 64), 128, 0, stream>>>(
      ybf, Wkb, nullptr, nullptr, Kbf, NTOK, DM, DA, 1, 0);
  gemm_bf16_kernel<<<dim3(NTOK / 64, DM / 64), 128, 0, stream>>>(
      ybf, Wvb, nullptr, nullptr, Vbf, NTOK, DM, DM, 1, 0);
  gemm_bf16_kernel<<<dim3(NTOK / 64, DM / 64), 128, 0, stream>>>(
      ybf, Wgb, bg, nullptr, Gf, NTOK, DM, DM, 0, 1);  // sigmoid gate, f32

  // attention + gating -> ctx (bf16)
  attn_kernel<<<dim3(LL / 16, BB), 128, 0, stream>>>(Qb, Kbf, Vbf, am, Gf, ctx);

  // h = ctx @ Wout + x
  gemm_bf16_kernel<<<dim3(NTOK / 64, DM / 64), 128, 0, stream>>>(
      ctx, Wob, nullptr, x, hf, NTOK, DM, DM, 0, 0);

  // LN2 -> FFN
  ln_kernel<<<NTOK, 128, 0, stream>>>(hf, l2w, l2b, nullptr, hnb);
  gemm_bf16_kernel<<<dim3(NTOK / 64, DF / 64), 128, 0, stream>>>(
      hnb, W1b, b1, nullptr, t1b, NTOK, DM, DF, 1, 2);  // exact GELU
  gemm_bf16_kernel<<<dim3(NTOK / 64, DM / 64), 128, 0, stream>>>(
      t1b, W2b, b2, hf, d_out, NTOK, DF, DM, 0, 0);     // out = h + ffn
}
